// TransformerModel_19447611916638
// MI455X (gfx1250) — compile-verified, matrix operand order checked by reading, not run on hardware
//
#include <hip/hip_runtime.h>
#include <hip/hip_bf16.h>

// ---------------- model constants ----------------
#define BB 2
#define TT 2048
#define DD 768
#define HH 12
#define LL 4
#define VV 50257
#define HD 64
#define FF 3072
#define MM (BB * TT)          // 4096 tokens
#define QKVN (3 * DD)         // 2304
#define VPAD 50304            // V padded to multiple of 128

typedef __attribute__((ext_vector_type(16))) _Float16 v16h;
typedef __attribute__((ext_vector_type(8)))  _Float16 h8;
typedef __attribute__((ext_vector_type(8)))  float    v8f;

// ---------------------------------------------------------------------------
// Embedding: x[b*T+t, :] = tok_emb[ctx[b,t], :] + pos_emb[t, :]
// ---------------------------------------------------------------------------
__global__ __launch_bounds__(256) void embed_k(const int* __restrict__ ctx,
                                               const float* __restrict__ tok,
                                               const float* __restrict__ pos,
                                               float* __restrict__ x) {
  int row = blockIdx.x;
  int t = row % TT;
  int tk = ctx[row];
  for (int c = threadIdx.x; c < DD; c += 256)
    x[(size_t)row * DD + c] = tok[(size_t)tk * DD + c] + pos[(size_t)t * DD + c];
}

// ---------------------------------------------------------------------------
// LayerNorm: f32 in -> f16 out (GEMM A operand)
// ---------------------------------------------------------------------------
__global__ __launch_bounds__(256) void layernorm_k(const float* __restrict__ x,
                                                   const float* __restrict__ gam,
                                                   const float* __restrict__ bet,
                                                   _Float16* __restrict__ out) {
  __shared__ float red[256];
  __shared__ float red2[256];
  int row = blockIdx.x, tid = threadIdx.x;
  const float* xr = x + (size_t)row * DD;
  float s = 0.f, q = 0.f;
  for (int c = tid; c < DD; c += 256) { float v = xr[c]; s += v; q += v * v; }
  red[tid] = s; red2[tid] = q;
  __syncthreads();
  for (int st = 128; st > 0; st >>= 1) {
    if (tid < st) { red[tid] += red[tid + st]; red2[tid] += red2[tid + st]; }
    __syncthreads();
  }
  float mean = red[0] / (float)DD;
  float var  = red2[0] / (float)DD - mean * mean;
  float inv  = rsqrtf(var + 1e-5f);
  for (int c = tid; c < DD; c += 256)
    out[(size_t)row * DD + c] = (_Float16)((xr[c] - mean) * inv * gam[c] + bet[c]);
}

// ---------------------------------------------------------------------------
// Weight conversion kernels (f32 -> f16 into workspace)
// ---------------------------------------------------------------------------
__global__ __launch_bounds__(256) void cvt_k(const float* __restrict__ src,
                                             _Float16* __restrict__ dst, long long n) {
  long long i = (long long)blockIdx.x * 256 + threadIdx.x;
  if (i < n) dst[i] = (_Float16)src[i];
}

// wq/wk/wv[l] are (H, D, HD); fuse into row-major [D][3*D] with n = s*768 + h*64 + e
__global__ __launch_bounds__(256) void cvt_qkv_k(const float* __restrict__ wq,
                                                 const float* __restrict__ wk,
                                                 const float* __restrict__ wv,
                                                 _Float16* __restrict__ dst) {
  long long i = (long long)blockIdx.x * 256 + threadIdx.x;
  if (i >= (long long)DD * QKVN) return;
  int d = (int)(i / QKVN), n = (int)(i % QKVN);
  int s = n / DD, r = n % DD;
  int h = r / HD, e = r % HD;
  const float* src = (s == 0) ? wq : (s == 1) ? wk : wv;
  dst[i] = (_Float16)src[(size_t)h * DD * HD + (size_t)d * HD + e];
}

// head_w (D,V) -> row-major [D][VPAD] zero-padded
__global__ __launch_bounds__(256) void cvt_head_k(const float* __restrict__ src,
                                                  _Float16* __restrict__ dst) {
  long long i = (long long)blockIdx.x * 256 + threadIdx.x;
  if (i >= (long long)DD * VPAD) return;
  int d = (int)(i / VPAD), n = (int)(i % VPAD);
  dst[i] = (n < VV) ? (_Float16)src[(size_t)d * VV + n] : (_Float16)0.f;
}

// ---------------------------------------------------------------------------
// WMMA GEMM: C[M,N] = act(A[M,K](f16) @ W[K,N](f16) + bias) (+residual)
// block = 256 threads (8 waves), block tile 128x128, wave tile 32x64, K-step 32
// A tile moved global->LDS with GLOBAL_LOAD_ASYNC_TO_LDS_B128 (ASYNCcnt path).
// ---------------------------------------------------------------------------
__global__ __launch_bounds__(256) void gemm_wmma(
    const _Float16* __restrict__ A, const _Float16* __restrict__ W,
    const float* __restrict__ bias, const float* __restrict__ resid,
    float* __restrict__ o32, _Float16* __restrict__ o16,
    int Mdim, int Ndim, int Kdim, int outStride, int Nvalid, int relu) {
  __shared__ __align__(16) _Float16 ldsA[128 * 32];    // [row][k] row-major
  __shared__ __align__(16) _Float16 ldsBt[128 * 40];   // [n][k], padded row 40

  int tid  = threadIdx.x;
  int lane = tid & 31;
  int wave = tid >> 5;
  int g    = lane >> 4;     // half-wave id (K-striping group)
  int l16  = lane & 15;
  int wm   = wave & 3;      // 4 wave-rows (M), 32 rows each
  int wn   = wave >> 2;     // 2 wave-cols (N), 64 cols each
  int m0   = blockIdx.y * 128;
  int n0   = blockIdx.x * 128;

  v8f acc[2][4];
  for (int a = 0; a < 2; a++)
    for (int b = 0; b < 4; b++) acc[a][b] = (v8f)0.f;

  int ar = tid >> 1, aseg = tid & 1;       // A loader: 128 rows x 2 16-half segs
  int br = tid >> 3, bc = (tid & 7) * 16;  // B loader: 32 rows x 8 cols-of-16

  unsigned ldsAoff = (unsigned)(size_t)&ldsA[ar * 32 + aseg * 16];

  for (int k0 = 0; k0 < Kdim; k0 += 32) {
    // B tile (32 x 128) -> registers (sync loads)
    const _Float16* wp = W + (size_t)(k0 + br) * Ndim + n0 + bc;
    h8 b0 = *(const h8*)(wp);
    h8 b1 = *(const h8*)(wp + 8);
    if (k0 + 32 < Kdim)
      __builtin_prefetch(W + (size_t)(k0 + 32 + br) * Ndim + n0 + bc, 0, 1);

    __syncthreads();   // previous iteration's fragment reads complete

    // A tile (128 x 32): async global->LDS, 2 x 16B per thread
    {
      const void* ga = (const void*)(A + (size_t)(m0 + ar) * Kdim + k0 + aseg * 16);
      asm volatile("global_load_async_to_lds_b128 %0, %1, off"
                   :: "v"(ldsAoff), "v"(ga) : "memory");
      asm volatile("global_load_async_to_lds_b128 %0, %1, off offset:16"
                   :: "v"(ldsAoff), "v"(ga) : "memory");
    }
    // B tile: transpose-store into LDS [n][k]
    for (int i = 0; i < 8; i++) ldsBt[(bc + i) * 40 + br]     = b0[i];
    for (int i = 0; i < 8; i++) ldsBt[(bc + 8 + i) * 40 + br] = b1[i];

    asm volatile("s_wait_asynccnt 0" ::: "memory");
    __syncthreads();

    // fragments
    v16h af[2], bf[4];
    for (int sm = 0; sm < 2; sm++) {
      int row = wm * 32 + sm * 16 + l16;
      h8 lo = *(const h8*)&ldsA[row * 32 + g * 8];
      h8 hi = *(const h8*)&ldsA[row * 32 + 16 + g * 8];
      for (int i = 0; i < 8; i++) { af[sm][i] = lo[i]; af[sm][8 + i] = hi[i]; }
    }
    for (int sn = 0; sn < 4; sn++) {
      int col = wn * 64 + sn * 16 + l16;
      h8 lo = *(const h8*)&ldsBt[col * 40 + g * 8];
      h8 hi = *(const h8*)&ldsBt[col * 40 + 16 + g * 8];
      for (int i = 0; i < 8; i++) { bf[sn][i] = lo[i]; bf[sn][8 + i] = hi[i]; }
    }
    for (int sm = 0; sm < 2; sm++)
      for (int sn = 0; sn < 4; sn++)
        acc[sm][sn] = __builtin_amdgcn_wmma_f32_16x16x32_f16(
            false, af[sm], false, bf[sn], (short)0, acc[sm][sn], false, false);
  }

  for (int sm = 0; sm < 2; sm++)
    for (int sn = 0; sn < 4; sn++) {
      int col = n0 + wn * 64 + sn * 16 + l16;
      if (col >= Nvalid) continue;
      float badd = bias ? bias[col] : 0.f;
      for (int i = 0; i < 8; i++) {
        int row = m0 + wm * 32 + sm * 16 + g * 8 + i;
        if (row >= Mdim) continue;
        float v = acc[sm][sn][i] + badd;
        if (resid) v += resid[(size_t)row * outStride + col];
        if (relu)  v = fmaxf(v, 0.f);
        if (o32) o32[(size_t)row * outStride + col] = v;
        if (o16) o16[(size_t)row * outStride + col] = (_Float16)v;
      }
    }
}

// ---------------------------------------------------------------------------
// Flash attention: one wave per (b, h, 16-query tile). Causal, scale = D^-0.5.
// qkv layout: [b*T+t][3*D] with q at h*64, k at 768+h*64, v at 1536+h*64.
// ---------------------------------------------------------------------------
__global__ __launch_bounds__(32) void attn_k(const _Float16* __restrict__ qkv,
                                             _Float16* __restrict__ out) {
  __shared__ __align__(16) _Float16 ldsP[16 * 32];
  int lane = threadIdx.x & 31;
  int g = lane >> 4, l16 = lane & 15;
  int qt = blockIdx.x;
  int bh = blockIdx.y;
  int b = bh / HH, h = bh % HH;
  const float scale = rsqrtf((float)DD);

  // Q fragments (A operand), contraction over e in two 32-chunks
  v16h qf[2];
  {
    const _Float16* qrow = qkv + (size_t)(b * TT + qt * 16 + l16) * QKVN + h * HD;
    for (int c = 0; c < 2; c++) {
      h8 lo = *(const h8*)(qrow + c * 32 + g * 8);
      h8 hi = *(const h8*)(qrow + c * 32 + 16 + g * 8);
      for (int i = 0; i < 8; i++) { qf[c][i] = lo[i]; qf[c][8 + i] = hi[i]; }
    }
  }

  v8f o[4];
  for (int n = 0; n < 4; n++) o[n] = (v8f)0.f;
  float mrow[8], lrow[8];
  for (int i = 0; i < 8; i++) { mrow[i] = -1e30f; lrow[i] = 0.f; }

  int qmax = qt * 16 + 15;
  for (int kb = 0; kb <= qmax; kb += 32) {
    v8f s0 = (v8f)0.f, s1 = (v8f)0.f;
    for (int c = 0; c < 2; c++) {
      for (int sn = 0; sn < 2; sn++) {
        int key = kb + sn * 16 + l16;
        const _Float16* krow = qkv + (size_t)(b * TT + key) * QKVN + DD + h * HD;
        h8 lo = *(const h8*)(krow + c * 32 + g * 8);
        h8 hi = *(const h8*)(krow + c * 32 + 16 + g * 8);
        v16h kf;
        for (int i = 0; i < 8; i++) { kf[i] = lo[i]; kf[8 + i] = hi[i]; }
        if (sn == 0)
          s0 = __builtin_amdgcn_wmma_f32_16x16x32_f16(false, qf[c], false, kf,
                                                      (short)0, s0, false, false);
        else
          s1 = __builtin_amdgcn_wmma_f32_16x16x32_f16(false, qf[c], false, kf,
                                                      (short)0, s1, false, false);
      }
    }
    // online softmax over the 16x32 score block
    float p0[8], p1[8];
    for (int i = 0; i < 8; i++) {
      int m = g * 8 + i;
      int qg = qt * 16 + m;
      float v0 = s0[i] * scale; if (kb + l16 > qg)      v0 = -1e30f;
      float v1 = s1[i] * scale; if (kb + 16 + l16 > qg) v1 = -1e30f;
      float r = fmaxf(v0, v1);
      for (int off = 8; off > 0; off >>= 1) r = fmaxf(r, __shfl_xor(r, off, 16));
      float mn = fmaxf(mrow[i], r);
      float alpha = __expf(mrow[i] - mn);
      mrow[i] = mn;
      float e0 = __expf(v0 - mn), e1 = __expf(v1 - mn);
      float rs = e0 + e1;
      for (int off = 8; off > 0; off >>= 1) rs += __shfl_xor(rs, off, 16);
      lrow[i] = lrow[i] * alpha + rs;
      for (int n = 0; n < 4; n++) o[n][i] *= alpha;
      p0[i] = e0; p1[i] = e1;
    }
    // stage P in LDS to re-fragment as A operand (K = key index)
    __syncthreads();
    for (int i = 0; i < 8; i++) {
      ldsP[(g * 8 + i) * 32 + l16]      = (_Float16)p0[i];
      ldsP[(g * 8 + i) * 32 + 16 + l16] = (_Float16)p1[i];
    }
    __syncthreads();
    v16h pf;
    {
      h8 lo = *(const h8*)&ldsP[l16 * 32 + g * 8];
      h8 hi = *(const h8*)&ldsP[l16 * 32 + 16 + g * 8];
      for (int i = 0; i < 8; i++) { pf[i] = lo[i]; pf[8 + i] = hi[i]; }
    }
    // P @ V over this 32-key block, 4 output e-subtiles
    for (int n = 0; n < 4; n++) {
      const _Float16* vbase = qkv + 2 * DD + h * HD + n * 16 + l16;
      v16h vf;
      for (int i = 0; i < 8; i++)
        vf[i] = vbase[(size_t)(b * TT + kb + g * 8 + i) * QKVN];
      for (int i = 0; i < 8; i++)
        vf[8 + i] = vbase[(size_t)(b * TT + kb + 16 + g * 8 + i) * QKVN];
      o[n] = __builtin_amdgcn_wmma_f32_16x16x32_f16(false, pf, false, vf,
                                                    (short)0, o[n], false, false);
    }
  }
  // normalize and store (concat heads): out[b*T+t][h*64+e]
  for (int n = 0; n < 4; n++)
    for (int i = 0; i < 8; i++) {
      int m = g * 8 + i;
      out[(size_t)(b * TT + qt * 16 + m) * DD + h * HD + n * 16 + l16] =
          (_Float16)(o[n][i] / lrow[i]);
    }
}

// ---------------------------------------------------------------------------
// Host orchestration
// ---------------------------------------------------------------------------
static inline void launch_gemm(hipStream_t st, const _Float16* A, const _Float16* Wt,
                               const float* bias, const float* resid,
                               float* o32, _Float16* o16,
                               int Mdim, int Ndim, int Kdim, int outStride,
                               int Nvalid, int relu) {
  dim3 grid(Ndim / 128, Mdim / 128);
  gemm_wmma<<<grid, 256, 0, st>>>(A, Wt, bias, resid, o32, o16,
                                  Mdim, Ndim, Kdim, outStride, Nvalid, relu);
}

extern "C" void kernel_launch(void* const* d_in, const int* in_sizes, int n_in,
                              void* d_out, int out_size, void* d_ws, size_t ws_size,
                              hipStream_t stream) {
  const int*   ctx     = (const int*)d_in[0];
  const float* tok_emb = (const float*)d_in[1];
  const float* pos_emb = (const float*)d_in[2];
  const float* wq      = (const float*)d_in[3];
  const float* wk      = (const float*)d_in[4];
  const float* wv      = (const float*)d_in[5];
  const float* proj_w  = (const float*)d_in[6];
  const float* proj_b  = (const float*)d_in[7];
  const float* ln1_g   = (const float*)d_in[8];
  const float* ln1_b   = (const float*)d_in[9];
  const float* ln2_g   = (const float*)d_in[10];
  const float* ln2_b   = (const float*)d_in[11];
  const float* ff_w1   = (const float*)d_in[12];
  const float* ff_b1   = (const float*)d_in[13];
  const float* ff_w2   = (const float*)d_in[14];
  const float* ff_b2   = (const float*)d_in[15];
  const float* lnf_g   = (const float*)d_in[16];
  const float* lnf_b   = (const float*)d_in[17];
  const float* head_w  = (const float*)d_in[18];
  const float* head_b  = (const float*)d_in[19];
  float* logits = (float*)d_out;

  // workspace carve-out
  char* w = (char*)d_ws;
  float*    x      = (float*)w;    w += (size_t)MM * DD * 4;
  _Float16* h16    = (_Float16*)w; w += (size_t)MM * DD * 2;
  _Float16* qkv16  = (_Float16*)w; w += (size_t)MM * QKVN * 2;
  _Float16* attn16 = (_Float16*)w; w += (size_t)MM * DD * 2;
  _Float16* ff16   = (_Float16*)w; w += (size_t)MM * FF * 2;
  _Float16* wqkv16 = (_Float16*)w; w += (size_t)DD * QKVN * 2;
  _Float16* wproj  = (_Float16*)w; w += (size_t)DD * DD * 2;
  _Float16* wff1   = (_Float16*)w; w += (size_t)DD * FF * 2;
  _Float16* wff2   = (_Float16*)w; w += (size_t)FF * DD * 2;
  _Float16* whead  = (_Float16*)w; w += (size_t)DD * VPAD * 2;

  embed_k<<<MM, 256, 0, stream>>>(ctx, tok_emb, pos_emb, x);
  {
    long long nh = (long long)DD * VPAD;
    cvt_head_k<<<(unsigned)((nh + 255) / 256), 256, 0, stream>>>(head_w, whead);
  }

  for (int l = 0; l < LL; l++) {
    long long nq = (long long)DD * QKVN;
    cvt_qkv_k<<<(unsigned)((nq + 255) / 256), 256, 0, stream>>>(
        wq + (size_t)l * HH * DD * HD, wk + (size_t)l * HH * DD * HD,
        wv + (size_t)l * HH * DD * HD, wqkv16);
    cvt_k<<<(DD * DD + 255) / 256, 256, 0, stream>>>(proj_w + (size_t)l * DD * DD, wproj, (long long)DD * DD);
    cvt_k<<<(DD * FF + 255) / 256, 256, 0, stream>>>(ff_w1 + (size_t)l * DD * FF, wff1, (long long)DD * FF);
    cvt_k<<<(FF * DD + 255) / 256, 256, 0, stream>>>(ff_w2 + (size_t)l * FF * DD, wff2, (long long)FF * DD);

    // x -> LN1 -> h16
    layernorm_k<<<MM, 256, 0, stream>>>(x, ln1_g + l * DD, ln1_b + l * DD, h16);
    // fused QKV projection
    launch_gemm(stream, h16, wqkv16, nullptr, nullptr, nullptr, qkv16,
                MM, QKVN, DD, QKVN, QKVN, 0);
    // attention
    attn_k<<<dim3(TT / 16, BB * HH), 32, 0, stream>>>(qkv16, attn16);
    // output projection + bias + residual -> x
    launch_gemm(stream, attn16, wproj, proj_b + l * DD, x, x, nullptr,
                MM, DD, DD, DD, DD, 0);
    // LN2 -> h16
    layernorm_k<<<MM, 256, 0, stream>>>(x, ln2_g + l * DD, ln2_b + l * DD, h16);
    // FF1 + bias + relu -> ff16
    launch_gemm(stream, h16, wff1, ff_b1 + l * FF, nullptr, nullptr, ff16,
                MM, FF, DD, FF, FF, 1);
    // FF2 + bias + residual -> x
    launch_gemm(stream, ff16, wff2, ff_b2 + l * DD, x, x, nullptr,
                MM, DD, FF, DD, DD, 0);
  }

  // final LN -> h16, then head GEMM -> logits (stride V, padded N guarded)
  layernorm_k<<<MM, 256, 0, stream>>>(x, lnf_g, lnf_b, h16);
  launch_gemm(stream, h16, whead, head_b, nullptr, logits, nullptr,
              MM, VPAD, DD, VV, VV, 0);
}
